// SpatialAttention_51805895524870
// MI455X (gfx1250) — matrix operand order repaired and checked by reading, not compiled, not run on hardware
//
#include <hip/hip_runtime.h>
#include <hip/hip_bf16.h>
#include <math.h>

// ---------------------------------------------------------------------------
// SpatialAttention for MI455X (gfx1250): bf16 WMMA for both einsums, with
// wide per-wave output tiles so each A fragment feeds 6-8 WMMAs.
// ---------------------------------------------------------------------------

typedef __attribute__((ext_vector_type(16))) __bf16       v16bf;
typedef __attribute__((ext_vector_type(8)))  float        v8f;
typedef __attribute__((ext_vector_type(4)))  unsigned int uint4v;

#define BB   32      // batch
#define CC   273     // channels
#define TT   3000    // time
#define OO   270     // output heads
#define DD   2048    // pos dim (2 * 32^2)
#define CP   288     // channels padded (K pad for GEMM2, N pad for GEMM1)
#define OP   272     // heads padded to mult of 16
#define TP   3072    // time padded to mult of 128 (24 * 128)
#define OT   17      // o tiles of 16     (272/16)
#define CSUB 6       // c sub-tiles per wave in k_scores (6*16 = 96 wide)
#define CG   3       // c groups of 96    (288/96)
#define TSUB 8       // t sub-tiles per wave in k_out   (8*16 = 128 wide)
#define TGN  24      // t groups of 128   (3072/128)

union BfVec {
  v16bf  v;
  uint4v u[2];
};

__device__ __forceinline__ unsigned short f2bf(float f) {
  unsigned int u = __builtin_bit_cast(unsigned int, f);
  unsigned int r = u + 0x7fffu + ((u >> 16) & 1u);   // round-to-nearest-even
  return (unsigned short)(r >> 16);
}

// --------------------------- 1) Fourier embedding --------------------------
__global__ void k_embed(const float* __restrict__ layout,
                        unsigned short* __restrict__ emb) {
  int bc = blockIdx.x;                       // b*CC + c
  float x = layout[(size_t)bc * 2 + 0];
  float y = layout[(size_t)bc * 2 + 1];
  const float inv = 1.0f / 1.2f;             // width = 1 + 2*margin
  float px = (x + 0.1f) * inv;
  float py = (y + 0.1f) * inv;
  unsigned short* row = emb + (size_t)bc * DD;
  for (int idx = threadIdx.x; idx < 1024; idx += 256) {
    int i = idx >> 5, j = idx & 31;
    float ph = 6.283185307179586f * (px * (float)i + py * (float)j);
    float s, c;
    __sincosf(ph, &s, &c);
    row[idx]        = f2bf(s);
    row[1024 + idx] = f2bf(c);
  }
}

// --------------------------- 2) heads -> bf16 -------------------------------
__global__ void k_cvt(const float* __restrict__ in,
                      unsigned short* __restrict__ out, int n) {
  int i = blockIdx.x * 256 + threadIdx.x;
  if (i < n) out[i] = f2bf(in[i]);
}

// ------------------ 3) brain f32 [b,c,t] -> bf16 [b,t,c] padded -------------
__global__ void k_transpose(const float* __restrict__ brain,
                            unsigned short* __restrict__ bt) {
  __shared__ unsigned short tile[32][33];
  int blk = blockIdx.x;
  int tt = blk % (TP / 32); blk /= (TP / 32);
  int ct = blk % (CP / 32);
  int b  = blk / (CP / 32);
  int c = ct * 32 + threadIdx.y;
  int t = tt * 32 + threadIdx.x;
  float v = 0.0f;
  if (c < CC && t < TT) v = brain[((size_t)b * CC + c) * TT + t];
  tile[threadIdx.y][threadIdx.x] = f2bf(v);
  __syncthreads();
  int to = tt * 32 + threadIdx.y;
  int co = ct * 32 + threadIdx.x;
  bt[((size_t)b * TP + to) * CP + co] = tile[threadIdx.x][threadIdx.y];
}

// ------------------ 4) scores[b,o,c] = emb . heads  (bf16 WMMA) -------------
// One wave per 16(o) x 96(c) tile: 6 accumulators share one A fragment.
__global__ void k_scores(const unsigned short* __restrict__ emb,
                         const unsigned short* __restrict__ headsb,
                         float* __restrict__ scores) {
  int wave = (blockIdx.x * blockDim.x + threadIdx.x) >> 5;
  int lane = threadIdx.x & 31;
  if (wave >= BB * OT * CG) return;          // wave-uniform
  int cg = wave % CG;
  int ot = (wave / CG) % OT;
  int b  = wave / (CG * OT);
  int h = lane >> 4;                          // lane half
  int r = lane & 15;

  int om = ot * 16 + r; if (om > OO - 1) om = OO - 1;   // clamp reads
  const char* aBase = (const char*)(headsb + (size_t)om * DD);
  const char* bBase[CSUB];
#pragma unroll
  for (int s = 0; s < CSUB; ++s) {
    int cn = cg * 96 + s * 16 + r; if (cn > CC - 1) cn = CC - 1;
    bBase[s] = (const char*)(emb + ((size_t)b * CC + cn) * DD);
  }

  v8f acc[CSUB];
#pragma unroll
  for (int s = 0; s < CSUB; ++s)
    acc[s] = (v8f){0.f, 0.f, 0.f, 0.f, 0.f, 0.f, 0.f, 0.f};

  for (int k0 = 0; k0 < DD; k0 += 32) {
    BfVec a;
    // A 16x32 bf16 layout: lane half h -> K {h*8..h*8+7, 16+h*8..16+h*8+7}
    a.u[0] = *(const uint4v*)(aBase + 2 * (k0 + h * 8));
    a.u[1] = *(const uint4v*)(aBase + 2 * (k0 + 16 + h * 8));
#pragma unroll
    for (int s = 0; s < CSUB; ++s) {
      BfVec bm;
      // B 32x16 bf16 layout: lane half h -> K {h*8..h*8+7, 16+h*8..16+h*8+7}
      bm.u[0] = *(const uint4v*)(bBase[s] + 2 * (k0 + h * 8));
      bm.u[1] = *(const uint4v*)(bBase[s] + 2 * (k0 + 16 + h * 8));
      acc[s] = __builtin_amdgcn_wmma_f32_16x16x32_bf16(
          false, a.v, false, bm.v, (short)0, acc[s], false, false);
    }
  }
  // D layout: VGPR v -> row m = v + 8*h, lane -> col n = r. Padded store.
  float* dbase = scores + ((size_t)b * OP + ot * 16) * CP + cg * 96;
#pragma unroll
  for (int s = 0; s < CSUB; ++s)
#pragma unroll
    for (int v = 0; v < 8; ++v)
      dbase[(size_t)(v + 8 * h) * CP + s * 16 + r] = acc[s][v];
}

// ------------------ 5) softmax over channels -> bf16 weights ----------------
__global__ void k_softmax(const float* __restrict__ scores,
                          unsigned short* __restrict__ wts) {
  __shared__ float red[128];
  int b = blockIdx.x / OP;
  int o = blockIdx.x % OP;
  const float* srow = scores + ((size_t)b * OP + o) * CP;
  unsigned short* wrow = wts + ((size_t)b * OP + o) * CP;
  int tid = threadIdx.x;

  if (o >= OO) {                              // padded rows: zeros (uniform)
    for (int c = tid; c < CP; c += 128) wrow[c] = 0;
    return;
  }
  float m = -3.4e38f;
  for (int c = tid; c < CC; c += 128) m = fmaxf(m, srow[c]);
  red[tid] = m; __syncthreads();
  for (int s = 64; s > 0; s >>= 1) {
    if (tid < s) red[tid] = fmaxf(red[tid], red[tid + s]);
    __syncthreads();
  }
  float mx = red[0]; __syncthreads();

  float sum = 0.0f;
  for (int c = tid; c < CC; c += 128) sum += __expf(srow[c] - mx);
  red[tid] = sum; __syncthreads();
  for (int s = 64; s > 0; s >>= 1) {
    if (tid < s) red[tid] += red[tid + s];
    __syncthreads();
  }
  float invTot = 1.0f / red[0];
  for (int c = tid; c < CP; c += 128) {
    unsigned short w = 0;
    if (c < CC) w = f2bf(__expf(srow[c] - mx) * invTot);
    wrow[c] = w;                              // zero K-padding for GEMM2
  }
}

// ------------------ 6) out[b,o,t] = weights @ brain  (bf16 WMMA) ------------
// One wave per 16(o) x 128(t) tile: 8 accumulators share one A fragment.
// K = 288 in 9 steps of 32.
__global__ void k_out(const unsigned short* __restrict__ wts,
                      const unsigned short* __restrict__ bt,
                      float* __restrict__ out) {
  int wave = (blockIdx.x * blockDim.x + threadIdx.x) >> 5;
  int lane = threadIdx.x & 31;
  if (wave >= BB * OT * TGN) return;          // wave-uniform
  int tg = wave % TGN;
  int ot = (wave / TGN) % OT;
  int b  = wave / (TGN * OT);
  int h = lane >> 4;
  int r = lane & 15;

  const char* aBase = (const char*)(wts + ((size_t)b * OP + ot * 16 + r) * CP);
  const char* bBase[TSUB];
#pragma unroll
  for (int nn = 0; nn < TSUB; ++nn)
    bBase[nn] = (const char*)(bt + ((size_t)b * TP + tg * 128 + nn * 16 + r) * CP);

  v8f acc[TSUB];
#pragma unroll
  for (int nn = 0; nn < TSUB; ++nn)
    acc[nn] = (v8f){0.f, 0.f, 0.f, 0.f, 0.f, 0.f, 0.f, 0.f};

  for (int k0 = 0; k0 < CP; k0 += 32) {
    BfVec a;
    a.u[0] = *(const uint4v*)(aBase + 2 * (k0 + h * 8));
    a.u[1] = *(const uint4v*)(aBase + 2 * (k0 + 16 + h * 8));
#pragma unroll
    for (int nn = 0; nn < TSUB; ++nn) {
      BfVec bm;
      bm.u[0] = *(const uint4v*)(bBase[nn] + 2 * (k0 + h * 8));
      bm.u[1] = *(const uint4v*)(bBase[nn] + 2 * (k0 + 16 + h * 8));
      acc[nn] = __builtin_amdgcn_wmma_f32_16x16x32_bf16(
          false, a.v, false, bm.v, (short)0, acc[nn], false, false);
    }
  }
#pragma unroll
  for (int nn = 0; nn < TSUB; ++nn) {
#pragma unroll
    for (int v = 0; v < 8; ++v) {
      int m = ot * 16 + v + 8 * h;
      int t = tg * 128 + nn * 16 + r;
      if (m < OO && t < TT)
        out[((size_t)b * OO + m) * TT + t] = acc[nn][v];
    }
  }
}

// ---------------------------------------------------------------------------
extern "C" void kernel_launch(void* const* d_in, const int* in_sizes, int n_in,
                              void* d_out, int out_size, void* d_ws, size_t ws_size,
                              hipStream_t stream) {
  (void)in_sizes; (void)n_in; (void)out_size; (void)ws_size;
  const float* brain  = (const float*)d_in[0];   // [32,273,3000]
  const float* layout = (const float*)d_in[1];   // [32,273,2]
  const float* heads  = (const float*)d_in[2];   // [270,2048]
  float* out = (float*)d_out;                    // [32,270,3000]

  char* ws = (char*)d_ws;
  size_t off = 0;
  unsigned short* emb    = (unsigned short*)(ws + off); off += (size_t)BB * CC * DD * 2; // 35.8 MB
  unsigned short* headsb = (unsigned short*)(ws + off); off += (size_t)OO * DD * 2;      //  1.1 MB
  float*          scores = (float*)         (ws + off); off += (size_t)BB * OP * CP * 4; // 10.0 MB
  unsigned short* wts    = (unsigned short*)(ws + off); off += (size_t)BB * OP * CP * 2; //  5.0 MB
  unsigned short* bt     = (unsigned short*)(ws + off); off += (size_t)BB * TP * CP * 2; // 56.6 MB

  // 1) fourier embedding -> bf16
  k_embed<<<BB * CC, 256, 0, stream>>>(layout, emb);
  // 2) heads -> bf16
  int nh = OO * DD;
  k_cvt<<<(nh + 255) / 256, 256, 0, stream>>>(heads, headsb, nh);
  // 3) brain [c][t] f32 -> [t][c] bf16 (padded, zero-filled)
  k_transpose<<<BB * (CP / 32) * (TP / 32), dim3(32, 32), 0, stream>>>(brain, bt);
  // 4) scores GEMM (WMMA bf16), 16x96 tiles per wave
  k_scores<<<(BB * OT * CG * 32 + 255) / 256, 256, 0, stream>>>(emb, headsb, scores);
  // 5) softmax over channels -> bf16 weights (zero-padded)
  k_softmax<<<BB * OP, 128, 0, stream>>>(scores, wts);
  // 6) out GEMM (WMMA bf16), 16x128 tiles per wave
  k_out<<<(BB * OT * TGN * 32 + 255) / 256, 256, 0, stream>>>(wts, bt, out);
}
